// Hyper_ViT_54125177864656
// MI455X (gfx1250) — matrix operand order repaired and checked by reading, not compiled, not run on hardware
//
#include <hip/hip_runtime.h>

// ============================================================================
// Hyper-ViT on MI455X (gfx1250): all GEMMs via v_wmma_f32_16x16x32_bf16.
// Roofline: ~250 GFLOP vs ~150MB traffic @23.3TB/s -> matrix-core bound, so
// bf16 WMMA with f32 accumulate everywhere. Attention is single-pass online
// flash (no NxN materialization, no score recompute): per 32-col chunk,
// QK^T via WMMA -> cross-lane row-max -> rescale O -> P through LDS to
// convert C/D layout -> A-fragment layout -> PV via WMMA against pre-
// transposed V. All masking is branchless (v_cndmask, no exec-mask churn).
// ============================================================================

typedef __bf16 bf16;
typedef __attribute__((ext_vector_type(16))) __bf16 v16bf;
typedef __attribute__((ext_vector_type(8)))  __bf16 v8bf;
typedef __attribute__((ext_vector_type(8)))  float  v8f;

constexpr int kDepth = 4;
constexpr int kC    = 768;
constexpr int kC3   = 2304;
constexpr int kNH   = 4;
constexpr int kHD   = 192;
constexpr int kCF   = 20;
constexpr int kB    = 4;
constexpr int kNSEQ = 1601;
constexpr int kNPAD = 1632;        // 51 * 32  (pads 1601)
constexpr int kBN   = kB * kNPAD;  // 6528 = 51 * 128
constexpr float kEps = 1e-5f;
constexpr float kNegInf = -3.0e38f;

// ---------------------------------------------------------------------------
// WMMA helpers (ISA 7.12.2 VGPR layouts)
// ---------------------------------------------------------------------------
__device__ inline v8f wmma_bf16(v16bf a, v16bf b, v8f c) {
  return __builtin_amdgcn_wmma_f32_16x16x32_bf16(false, a, false, b,
                                                 (short)0, c, false, false);
}

// A fragment, 16x32 bf16, row-major source (lda in elements).
// lanes 0-15: row=lane,   K = {k0..k0+7, k0+16..k0+23}
// lanes16-31: row=lane-16,K = {k0+8..k0+15, k0+24..k0+31}
__device__ inline v16bf load_frag_A(const bf16* __restrict__ A, int lda,
                                    int m0, int k0) {
  int lane = threadIdx.x & 31;
  const bf16* p = A + (size_t)(m0 + (lane & 15)) * lda + k0 + ((lane >> 4) * 8);
  v16bf f;
  *(v8bf*)&f       = *(const v8bf*)p;
  *((v8bf*)&f + 1) = *(const v8bf*)(p + 16);
  return f;
}

// B fragment, 32x16 bf16, from W stored row-major [Nout][K]:
// B[k][n] = W[n0+lane%16][k0 + (lane/16)*8 + {0..7, 16..23}] (matches A).
__device__ inline v16bf load_frag_B(const bf16* __restrict__ W, int ldw,
                                    int n0, int k0) {
  int lane = threadIdx.x & 31;
  const bf16* p = W + (size_t)(n0 + (lane & 15)) * ldw + k0 + ((lane >> 4) * 8);
  v16bf f;
  *(v8bf*)&f       = *(const v8bf*)p;
  *((v8bf*)&f + 1) = *(const v8bf*)(p + 16);
  return f;
}

__device__ inline float blk_sum(float v, float* tmp) {
  #pragma unroll
  for (int off = 16; off > 0; off >>= 1) v += __shfl_xor(v, off, 32);
  __syncthreads();
  if ((threadIdx.x & 31) == 0) tmp[threadIdx.x >> 5] = v;
  __syncthreads();
  float s = 0.f;
  int nw = (int)(blockDim.x >> 5);
  for (int i = 0; i < nw; ++i) s += tmp[i];
  return s;
}

// ---------------------------------------------------------------------------
// Generic WMMA GEMM:  out[M,N] = A[M,K] * W[N,K]^T  (+bias) (+residual accum)
// block = 256 thr = 8 waves; block tile 128x64; wave tile 16x64.
// All dims exactly divisible (kBN=51*128; N in {2304,768,192}; K in {768,192}).
// ---------------------------------------------------------------------------
template <int OUT_BF16, int BIAS, int ACCUM>
__global__ __launch_bounds__(256) void gemm_bf16(
    const bf16* __restrict__ A, int lda,
    const bf16* __restrict__ W, int ldw,
    void* __restrict__ outv, int ldo,
    const float* __restrict__ bias, int K) {
  int wave = threadIdx.x >> 5;
  int lane = threadIdx.x & 31;
  int m0 = blockIdx.x * 128 + wave * 16;
  int n0 = blockIdx.y * 64;
  v8f acc[4] = {};
  for (int k0 = 0; k0 < K; k0 += 32) {
    if (k0 + 32 < K)  // global_prefetch path
      __builtin_prefetch(A + (size_t)(m0 + (lane & 15)) * lda + k0 + 32, 0, 0);
    v16bf a = load_frag_A(A, lda, m0, k0);
    #pragma unroll
    for (int j = 0; j < 4; ++j) {
      v16bf b = load_frag_B(W, ldw, n0 + j * 16, k0);
      acc[j] = wmma_bf16(a, b, acc[j]);
    }
  }
  int half = lane >> 4, coll = lane & 15;
  #pragma unroll
  for (int j = 0; j < 4; ++j) {
    int col = n0 + j * 16 + coll;
    float bcol = BIAS ? bias[col] : 0.f;
    #pragma unroll
    for (int r = 0; r < 8; ++r) {
      size_t off = (size_t)(m0 + r + half * 8) * ldo + col;
      float v = acc[j][r] + bcol;
      if (OUT_BF16)      ((bf16*)outv)[off] = (bf16)v;
      else if (ACCUM)    ((float*)outv)[off] += v;
      else               ((float*)outv)[off] = v;
    }
  }
}

// ---------------------------------------------------------------------------
// Small elementwise / conversion kernels
// ---------------------------------------------------------------------------
__global__ void f32_to_bf16(const float* __restrict__ s, bf16* __restrict__ d,
                            int n) {
  int i = blockIdx.x * 256 + threadIdx.x;
  if (i < n) d[i] = (bf16)s[i];
}

// Mt[l][e][d] = M[l][d][e]  (so qM = q @ M == q @ Mt^T fits GEMM form)
__global__ void make_Mt(const float* __restrict__ M, bf16* __restrict__ Mt) {
  int i = blockIdx.x * 256 + threadIdx.x;
  if (i >= kDepth * kHD * kHD) return;
  int dd = i % kHD;
  int rr = i / kHD;
  int e  = rr % kHD;
  int l  = rr / kHD;
  Mt[i] = (bf16)M[((size_t)l * kHD + dd) * kHD + e];
}

__global__ void centers_sq(const float* __restrict__ centers,
                           float* __restrict__ ccsq) {
  int i = blockIdx.x;  // l*kCF + f
  __shared__ float tmp[8];
  const float* c = centers + (size_t)i * kC;
  float s = 0.f;
  #pragma unroll
  for (int j = 0; j < 3; ++j) {
    float v = c[threadIdx.x + j * 256];
    s += v * v;
  }
  float tot = blk_sum(s, tmp);
  if (threadIdx.x == 0) ccsq[i] = tot;
}

// ---------------------------------------------------------------------------
// Patch embed: conv(stride=4) as per-patch linear + cls + pos embed
// grid = kBN blocks (one row), 256 threads x 3 channels
// ---------------------------------------------------------------------------
__global__ void patch_embed(const float* __restrict__ x,
                            const float* __restrict__ cw,
                            const float* __restrict__ cb,
                            const float* __restrict__ cls,
                            const float* __restrict__ pos,
                            float* __restrict__ t) {
  int row = blockIdx.x;
  int b = row / kNPAD, n = row % kNPAD;
  float* o = t + (size_t)row * kC;
  #pragma unroll
  for (int j = 0; j < 3; ++j) {
    int c = threadIdx.x + j * 256;
    float r;
    if (n >= kNSEQ) {
      r = 0.f;
    } else if (n == 0) {
      r = cls[c] + pos[c];
    } else {
      int pi = n - 1, gy = pi / 40, gx = pi % 40;
      float acc = cb[c];
      const float* wr = cw + (size_t)c * 48;
      for (int ch = 0; ch < 3; ++ch)
        for (int py = 0; py < 4; ++py)
          for (int px = 0; px < 4; ++px)
            acc += x[(((size_t)b * 3 + ch) * 160 + gy * 4 + py) * 160 +
                     gx * 4 + px] * wr[ch * 16 + py * 4 + px];
      r = acc + pos[(size_t)n * kC + c];
    }
    o[c] = r;
  }
}

// ---------------------------------------------------------------------------
// LayerNorm -> bf16 (zeros padded rows so q/k/v of pad rows are 0)
// ---------------------------------------------------------------------------
__global__ void ln_to_bf16_kernel(const float* __restrict__ t,
                                  const float* __restrict__ w,
                                  const float* __restrict__ bb,
                                  bf16* __restrict__ out) {
  int row = blockIdx.x;
  int n = row % kNPAD;
  const float* x = t + (size_t)row * kC;
  bf16* o = out + (size_t)row * kC;
  __shared__ float tmp[8];
  bool live = (n < kNSEQ);
  float v[3] = {0.f, 0.f, 0.f};
  if (live) {
    #pragma unroll
    for (int j = 0; j < 3; ++j) v[j] = x[threadIdx.x + j * 256];
  }
  float mean = blk_sum(v[0] + v[1] + v[2], tmp) * (1.0f / kC);
  float d0 = v[0] - mean, d1 = v[1] - mean, d2 = v[2] - mean;
  float var = blk_sum(d0 * d0 + d1 * d1 + d2 * d2, tmp) * (1.0f / kC);
  float rstd = rsqrtf(var + kEps);
  #pragma unroll
  for (int j = 0; j < 3; ++j) {
    int c = threadIdx.x + j * 256;
    o[c] = live ? (bf16)((v[j] - mean) * rstd * w[c] + bb[c]) : (bf16)0.f;
  }
}

// ---------------------------------------------------------------------------
// qMq[idx] = q.qM, kMk[idx] = k.kM   (idx = (h*kB+b)*kNPAD + n)
// ---------------------------------------------------------------------------
__global__ void qk_dots(const bf16* __restrict__ qkv,
                        const bf16* __restrict__ qM,
                        const bf16* __restrict__ kM,
                        float* __restrict__ qMq, float* __restrict__ kMk) {
  int idx = blockIdx.x;
  int h = idx / kBN;
  int rem = idx - h * kBN;  // b*kNPAD + n
  int d = threadIdx.x;
  __shared__ float tmp[6];
  float qv = (float)qkv[(size_t)rem * kC3 + h * kHD + d] *
             (float)qM[(size_t)idx * kHD + d];
  float kv = (float)qkv[(size_t)rem * kC3 + kC + h * kHD + d] *
             (float)kM[(size_t)idx * kHD + d];
  float qs = blk_sum(qv, tmp);
  float ks = blk_sum(kv, tmp);
  if (threadIdx.x == 0) {
    qMq[idx] = qs;
    kMk[idx] = ks;
  }
}

// vT[((b*kNH+h)*kHD + d)*kNPAD + n] = v[b][n][h][d]
__global__ void transpose_v(const bf16* __restrict__ qkv,
                            bf16* __restrict__ vT) {
  size_t i = (size_t)blockIdx.x * 256 + threadIdx.x;  // exact grid
  int n = (int)(i % kNPAD);
  size_t r = i / kNPAD;
  int d = (int)(r % kHD); r /= kHD;
  int h = (int)(r % kNH);
  int b = (int)(r / kNH);
  vT[i] = qkv[((size_t)b * kNPAD + n) * kC3 + 2 * kC + h * kHD + d];
}

// ---------------------------------------------------------------------------
// Single-pass online flash Mahalanobis attention.
// grid = (17, B*NH), block = 192 thr = 6 waves -> exactly 102 row-tiles/(b,h).
// Per 32-col chunk: 12 WMMA scores, branchless mask, cross-lane row max
// (rows live across the 16 lanes of a half in C/D layout), rescale O by
// exp(m_old - m_new), P -> LDS -> A-fragment, 12 WMMA PV accumulate.
// ---------------------------------------------------------------------------
__global__ __launch_bounds__(192) void attn_kernel(
    const bf16* __restrict__ qM, const bf16* __restrict__ qkv,
    const bf16* __restrict__ vT, const float* __restrict__ qMq,
    const float* __restrict__ kMk, const float* __restrict__ scale_arr,
    int layer, bf16* __restrict__ y) {
  int wave = threadIdx.x >> 5;
  int lane = threadIdx.x & 31;
  int rt = blockIdx.x * 6 + wave;  // 0..101, exact
  int bh = blockIdx.y;
  int b = bh >> 2, h = bh & 3;
  int m0 = rt * 16;
  const bf16* qMh = qM + ((size_t)(h * kB + b)) * kNPAD * kHD;
  const bf16* kh  = qkv + (size_t)b * kNPAD * kC3 + kC + h * kHD;
  const bf16* vTh = vT + ((size_t)(b * kNH + h)) * kHD * kNPAD;
  const float* qq = qMq + (size_t)(h * kB + b) * kNPAD;
  const float* kk = kMk + (size_t)(h * kB + b) * kNPAD;
  float sc = scale_arr[layer];

  v16bf aq[6];
  #pragma unroll
  for (int kc = 0; kc < 6; ++kc) aq[kc] = load_frag_A(qMh, kHD, m0, kc * 32);

  int half = lane >> 4, coll = lane & 15;
  float qrow[8];
  #pragma unroll
  for (int r = 0; r < 8; ++r) qrow[r] = qq[m0 + r + half * 8];

  float mrow[8], lsum[8];
  #pragma unroll
  for (int r = 0; r < 8; ++r) { mrow[r] = kNegInf; lsum[r] = 0.f; }
  v8f O[12] = {};
  __shared__ __align__(16) bf16 plds[6][16 * 32];  // per-wave 16x32 P tile

  for (int n0 = 0; n0 < kNPAD; n0 += 32) {
    // ---- scores for two 16-wide tiles (12 WMMAs) --------------------------
    v8f s0 = {}, s1 = {};
    #pragma unroll
    for (int kc = 0; kc < 6; ++kc) {
      v16bf b0 = load_frag_B(kh, kC3, n0, kc * 32);
      v16bf b1 = load_frag_B(kh, kC3, n0 + 16, kc * 32);
      s0 = wmma_bf16(aq[kc], b0, s0);
      s1 = wmma_bf16(aq[kc], b1, s1);
    }
    int c0 = n0 + coll, c1 = n0 + 16 + coll;
    float kk0 = kk[c0], kk1 = kk[c1];
    bool ok0 = c0 < kNSEQ, ok1 = c1 < kNSEQ;
    float l0[8], l1[8], cm[8];
    #pragma unroll
    for (int r = 0; r < 8; ++r) {  // branchless mask -> v_cndmask
      l0[r] = ok0 ? -sc * (qrow[r] + kk0 - 2.0f * s0[r]) : kNegInf;
      l1[r] = ok1 ? -sc * (qrow[r] + kk1 - 2.0f * s1[r]) : kNegInf;
      cm[r] = fmaxf(l0[r], l1[r]);
    }
    // consistent row max across the 16 lanes of this half
    #pragma unroll
    for (int off = 1; off < 16; off <<= 1) {
      #pragma unroll
      for (int r = 0; r < 8; ++r)
        cm[r] = fmaxf(cm[r], __shfl_xor(cm[r], off, 32));
    }
    float corr[8];
    #pragma unroll
    for (int r = 0; r < 8; ++r) {
      float mn = fmaxf(mrow[r], cm[r]);
      corr[r] = __expf(mrow[r] - mn);  // first chunk: exp(-huge) -> 0
      mrow[r] = mn;
    }
    // P values (masked lanes underflow to 0), per-lane lsum partials
    #pragma unroll
    for (int r = 0; r < 8; ++r) {
      float p0 = __expf(l0[r] - mrow[r]);
      float p1 = __expf(l1[r] - mrow[r]);
      lsum[r] = lsum[r] * corr[r] + p0 + p1;
      plds[wave][(r + half * 8) * 32 + coll]      = (bf16)p0;
      plds[wave][(r + half * 8) * 32 + 16 + coll] = (bf16)p1;
    }
    // rescale O before accumulating this chunk
    #pragma unroll
    for (int dc = 0; dc < 12; ++dc) {
      #pragma unroll
      for (int r = 0; r < 8; ++r) O[dc][r] *= corr[r];
    }
    __syncthreads();
    v16bf pf;  // read back as A-fragment (lda=32, m0=0, k0=0)
    {
      const bf16* p = &plds[wave][0] + coll * 32 + half * 8;
      *(v8bf*)&pf       = *(const v8bf*)p;
      *((v8bf*)&pf + 1) = *(const v8bf*)(p + 16);
    }
    __syncthreads();
    #pragma unroll
    for (int dc = 0; dc < 12; ++dc) {
      v16bf bv = load_frag_B(vTh, kNPAD, dc * 16, n0);  // cols=d, K=seq
      O[dc] = wmma_bf16(pf, bv, O[dc]);
    }
  }
  // final row-sum across the half, then normalize + store
  #pragma unroll
  for (int off = 1; off < 16; off <<= 1) {
    #pragma unroll
    for (int r = 0; r < 8; ++r) lsum[r] += __shfl_xor(lsum[r], off, 32);
  }
  float inv[8];
  #pragma unroll
  for (int r = 0; r < 8; ++r) inv[r] = 1.0f / lsum[r];
  #pragma unroll
  for (int dc = 0; dc < 12; ++dc) {
    #pragma unroll
    for (int r = 0; r < 8; ++r) {
      int n = m0 + r + half * 8;
      int ch = h * kHD + dc * 16 + coll;
      y[((size_t)b * kNPAD + n) * kC + ch] = (bf16)(O[dc][r] * inv[r]);
    }
  }
}

// ---------------------------------------------------------------------------
// RBF MLP: t += exp(-beta * (|h|^2 - 2 h.c + |c|^2)) @ fc_w^T  (h = LN2(t))
// grid = B*NSEQ rows, 256 threads
// ---------------------------------------------------------------------------
__global__ void rbf_mlp(float* __restrict__ t, const float* __restrict__ w,
                        const float* __restrict__ bb,
                        const float* __restrict__ centers,
                        const float* __restrict__ beta,
                        const float* __restrict__ fc_w,
                        const float* __restrict__ ccsq) {
  int row = blockIdx.x;
  int b = row / kNSEQ, n = row % kNSEQ;
  float* x = t + ((size_t)b * kNPAD + n) * kC;
  __shared__ float tmp[8];
  __shared__ float cd[kCF];
  __shared__ float rr[kCF];
  float v[3], hv[3];
  #pragma unroll
  for (int j = 0; j < 3; ++j) v[j] = x[threadIdx.x + j * 256];
  float mean = blk_sum(v[0] + v[1] + v[2], tmp) * (1.0f / kC);
  float d0 = v[0] - mean, d1 = v[1] - mean, d2 = v[2] - mean;
  float var = blk_sum(d0 * d0 + d1 * d1 + d2 * d2, tmp) * (1.0f / kC);
  float rstd = rsqrtf(var + kEps);
  float hh = 0.f;
  #pragma unroll
  for (int j = 0; j < 3; ++j) {
    int c = threadIdx.x + j * 256;
    hv[j] = (v[j] - mean) * rstd * w[c] + bb[c];
    hh += hv[j] * hv[j];
  }
  float pd[kCF];
  #pragma unroll
  for (int f = 0; f < kCF; ++f) pd[f] = 0.f;
  #pragma unroll
  for (int j = 0; j < 3; ++j) {
    int c = threadIdx.x + j * 256;
    for (int f = 0; f < kCF; ++f) pd[f] += hv[j] * centers[(size_t)f * kC + c];
  }
  if (threadIdx.x < kCF) cd[threadIdx.x] = 0.f;
  float hhs = blk_sum(hh, tmp);  // includes __syncthreads after cd init
  for (int f = 0; f < kCF; ++f) atomicAdd(&cd[f], pd[f]);
  __syncthreads();
  if (threadIdx.x < kCF) {
    float dist = hhs - 2.0f * cd[threadIdx.x] + ccsq[threadIdx.x];
    rr[threadIdx.x] = __expf(-beta[threadIdx.x] * dist);
  }
  __syncthreads();
  #pragma unroll
  for (int j = 0; j < 3; ++j) {
    int c = threadIdx.x + j * 256;
    float acc = 0.f;
    for (int f = 0; f < kCF; ++f) acc += rr[f] * fc_w[(size_t)c * kCF + f];
    x[c] = v[j] + acc;
  }
}

// final LN, CLS row only -> out[b][c]
__global__ void final_ln(const float* __restrict__ t,
                         const float* __restrict__ w,
                         const float* __restrict__ bb,
                         float* __restrict__ out) {
  int b = blockIdx.x;
  const float* xr = t + (size_t)b * kNPAD * kC;
  __shared__ float tmp[8];
  float v[3];
  #pragma unroll
  for (int j = 0; j < 3; ++j) v[j] = xr[threadIdx.x + j * 256];
  float mean = blk_sum(v[0] + v[1] + v[2], tmp) * (1.0f / kC);
  float d0 = v[0] - mean, d1 = v[1] - mean, d2 = v[2] - mean;
  float var = blk_sum(d0 * d0 + d1 * d1 + d2 * d2, tmp) * (1.0f / kC);
  float rstd = rsqrtf(var + kEps);
  #pragma unroll
  for (int j = 0; j < 3; ++j) {
    int c = threadIdx.x + j * 256;
    out[(size_t)b * kC + c] = (v[j] - mean) * rstd * w[c] + bb[c];
  }
}

// ---------------------------------------------------------------------------
// Host orchestration (graph-capture safe: only kernel launches on stream).
// Workspace use ~120 MB carved from d_ws.
// ---------------------------------------------------------------------------
extern "C" void kernel_launch(void* const* d_in, const int* in_sizes, int n_in,
                              void* d_out, int out_size, void* d_ws,
                              size_t ws_size, hipStream_t stream) {
  (void)in_sizes; (void)n_in; (void)out_size; (void)ws_size;
  const float* x       = (const float*)d_in[0];
  const float* conv_w  = (const float*)d_in[1];
  const float* conv_b  = (const float*)d_in[2];
  const float* cls     = (const float*)d_in[3];
  const float* pos     = (const float*)d_in[4];
  const float* ln1_w   = (const float*)d_in[5];
  const float* ln1_b   = (const float*)d_in[6];
  const float* qkv_w   = (const float*)d_in[7];
  const float* scale   = (const float*)d_in[8];
  const float* Mmat    = (const float*)d_in[9];
  const float* proj_w  = (const float*)d_in[10];
  const float* proj_b  = (const float*)d_in[11];
  const float* ln2_w   = (const float*)d_in[12];
  const float* ln2_b   = (const float*)d_in[13];
  const float* centers = (const float*)d_in[14];
  const float* beta    = (const float*)d_in[15];
  const float* fc_w    = (const float*)d_in[16];
  const float* lnf_w   = (const float*)d_in[17];
  const float* lnf_b   = (const float*)d_in[18];
  float* out = (float*)d_out;

  char* p = (char*)d_ws;
  auto carve = [&](size_t bytes) -> char* {
    char* r = p;
    p += (bytes + 255) & ~(size_t)255;
    return r;
  };
  float* t       = (float*)carve((size_t)kBN * kC * 4);
  bf16*  hb      = (bf16*) carve((size_t)kBN * kC * 2);
  bf16*  qkvb    = (bf16*) carve((size_t)kBN * kC3 * 2);
  bf16*  qMb     = (bf16*) carve((size_t)kNH * kBN * kHD * 2);
  bf16*  kMb     = (bf16*) carve((size_t)kNH * kBN * kHD * 2);
  float* qMq     = (float*)carve((size_t)kNH * kBN * 4);
  float* kMk     = (float*)carve((size_t)kNH * kBN * 4);
  bf16*  vT      = (bf16*) carve((size_t)kNH * kBN * kHD * 2);
  bf16*  yb      = (bf16*) carve((size_t)kBN * kC * 2);
  bf16*  qkvw_b  = (bf16*) carve((size_t)kDepth * kC3 * kC * 2);
  bf16*  projw_b = (bf16*) carve((size_t)kDepth * kC * kC * 2);
  bf16*  Mt      = (bf16*) carve((size_t)kDepth * kHD * kHD * 2);
  float* ccsq    = (float*)carve((size_t)kDepth * kCF * 4);

  // weight prep
  int nqkvw = kDepth * kC3 * kC;
  f32_to_bf16<<<(nqkvw + 255) / 256, 256, 0, stream>>>(qkv_w, qkvw_b, nqkvw);
  int nprojw = kDepth * kC * kC;
  f32_to_bf16<<<(nprojw + 255) / 256, 256, 0, stream>>>(proj_w, projw_b, nprojw);
  make_Mt<<<(kDepth * kHD * kHD + 255) / 256, 256, 0, stream>>>(Mmat, Mt);
  centers_sq<<<kDepth * kCF, 256, 0, stream>>>(centers, ccsq);

  patch_embed<<<kBN, 256, 0, stream>>>(x, conv_w, conv_b, cls, pos, t);

  for (int i = 0; i < kDepth; ++i) {
    ln_to_bf16_kernel<<<kBN, 256, 0, stream>>>(t, ln1_w + i * kC,
                                               ln1_b + i * kC, hb);
    gemm_bf16<1, 0, 0><<<dim3(kBN / 128, kC3 / 64), 256, 0, stream>>>(
        hb, kC, qkvw_b + (size_t)i * kC3 * kC, kC, qkvb, kC3, nullptr, kC);
    for (int h = 0; h < kNH; ++h) {
      gemm_bf16<1, 0, 0><<<dim3(kBN / 128, kHD / 64), 256, 0, stream>>>(
          qkvb + h * kHD, kC3, Mt + (size_t)i * kHD * kHD, kHD,
          qMb + (size_t)h * kBN * kHD, kHD, nullptr, kHD);
      gemm_bf16<1, 0, 0><<<dim3(kBN / 128, kHD / 64), 256, 0, stream>>>(
          qkvb + kC + h * kHD, kC3, Mt + (size_t)i * kHD * kHD, kHD,
          kMb + (size_t)h * kBN * kHD, kHD, nullptr, kHD);
    }
    qk_dots<<<kNH * kBN, 192, 0, stream>>>(qkvb, qMb, kMb, qMq, kMk);
    transpose_v<<<(kNH * kBN * kHD) / 256, 256, 0, stream>>>(qkvb, vT);
    attn_kernel<<<dim3(17, kB * kNH), 192, 0, stream>>>(qMb, qkvb, vT, qMq,
                                                        kMk, scale, i, yb);
    gemm_bf16<0, 1, 1><<<dim3(kBN / 128, kC / 64), 256, 0, stream>>>(
        yb, kC, projw_b + (size_t)i * kC * kC, kC, t, kC, proj_b + i * kC, kC);
    rbf_mlp<<<kB * kNSEQ, 256, 0, stream>>>(
        t, ln2_w + i * kC, ln2_b + i * kC, centers + (size_t)i * kCF * kC,
        beta + i * kCF, fc_w + (size_t)i * kC * kCF, ccsq + i * kCF);
  }
  final_ln<<<kB, 256, 0, stream>>>(t, lnf_w, lnf_b, out);
}
